// LengthRegulator_18820546691103
// MI455X (gfx1250) — matrix operand order repaired and checked by reading, not compiled, not run on hardware
//
#include <hip/hip_runtime.h>
#include <stdint.h>

#define B_    32
#define T_    1024
#define H_    256
#define K_    5
#define RED   1280      // H_*K_
#define TMEL  4096
#define TP    1028      // T_ + 4 (2-row halo each side)
#define ROWS  32        // token rows per GEMM block
#define KSTEPS 40       // RED / 32

typedef __attribute__((ext_vector_type(16))) __bf16 v16bf;
typedef __attribute__((ext_vector_type(8)))  float  v8f;

union BfVec { uint4 u[2]; v16bf v; };

__device__ __forceinline__ unsigned short f2bf(float f) {
    union { float f; unsigned u; } x; x.f = f;
    unsigned r = x.u + 0x7FFFu + ((x.u >> 16) & 1u);   // round-to-nearest-even
    return (unsigned short)(r >> 16);
}

// ---------------------------------------------------------------------------
// Repack conv2_w [H_out, H_in, K] (f32) -> Wr [H_out][k*H + i] (bf16) so that
// WMMA B fragments are contiguous 128-bit loads.
__global__ void pack_w(const float* __restrict__ w, unsigned short* __restrict__ wr) {
    int idx = blockIdx.x * 256 + threadIdx.x;          // < 256*1280
    int o = idx / RED, j = idx % RED;
    int ko = j >> 8, i = j & 255;                      // j = ko*256 + i
    wr[idx] = f2bf(w[o * RED + i * K_ + ko]);
}

// x [B,T,H] f32 -> xp [B, T+4, H] bf16 with 2 zero rows of halo each side.
__global__ void pack_x(const float* __restrict__ x, unsigned short* __restrict__ xp) {
    int idx = blockIdx.x * 256 + threadIdx.x;          // < B_*TP*H_ (8,421,376)
    int h  = idx & (H_ - 1);
    int r  = idx / H_;
    int tp = r % TP;
    int b  = r / TP;
    int t  = tp - 2;
    float v = (t >= 0 && t < T_) ? x[((size_t)b * T_ + t) * H_ + h] : 0.f;
    xp[idx] = f2bf(v);
}

// ---------------------------------------------------------------------------
// Fused conv1d(K=5) GEMM (bf16 WMMA, f32 accum) + bias + LayerNorm + ReLU +
// linear projection -> masked m[b,t].
__global__ void __launch_bounds__(256)
conv_ln_kernel(const unsigned short* __restrict__ xp,
               const unsigned short* __restrict__ wr,
               const float* __restrict__ bias,
               const float* __restrict__ g,
               const float* __restrict__ bet,
               const float* __restrict__ lw,
               const float* __restrict__ lb,
               const int*   __restrict__ token_nums,
               float*       __restrict__ m_out)
{
    extern __shared__ char smem[];
    unsigned short* As = (unsigned short*)smem;   // [ROWS][RED] bf16 (80 KB)
    float*          Cs = (float*)smem;            // [ROWS][H_] f32 (32 KB, reuses As)

    int tile = blockIdx.x;
    int b    = tile >> 5;                 // T_/ROWS == 32 blocks per batch
    int t0   = (tile & 31) * ROWS;
    int tid  = threadIdx.x;
    int lane = tid & 31;
    int wave = tid >> 5;

    // ---- stage A tile: ROWS rows, each 1280 contiguous bf16 from padded x
    {
        const unsigned int* src = (const unsigned int*)xp;
        unsigned int*       dst = (unsigned int*)As;
        #pragma unroll
        for (int it = 0; it < (ROWS * RED / 2) / 256; ++it) {
            int di  = it * 256 + tid;            // dword index into As
            int row = di / (RED / 2);
            int c2  = di % (RED / 2);
            dst[di] = src[(((size_t)b * TP + t0 + row) * H_ >> 1) + c2];
        }
    }
    __syncthreads();

    // wave handles 2 row-tiles x 2 col-tiles of 16x16
    v8f acc[2][2] = { { {}, {} }, { {}, {} } };
    int n0    = wave * 2;
    int alane = lane & 15;
    int kbA   = (lane >> 4) * 8;     // A: lanes 0-15 K 0-7/16-23, lanes 16-31 K 8-15/24-31
    int kbB   = (lane >> 4) * 8;     // B: lanes 0-15 K 0-7/16-23, lanes 16-31 K 8-15/24-31

    for (int s = 0; s < KSTEPS; ++s) {
        int k0 = s * 32;
        BfVec aF[2];
        #pragma unroll
        for (int rt = 0; rt < 2; ++rt) {
            int row = rt * 16 + alane;
            aF[rt].u[0] = *(const uint4*)(As + row * RED + k0 + kbA);
            aF[rt].u[1] = *(const uint4*)(As + row * RED + k0 + 16 + kbA);
        }
        BfVec bF[2];
        #pragma unroll
        for (int ct = 0; ct < 2; ++ct) {
            int o = (n0 + ct) * 16 + alane;
            const uint4* p = (const uint4*)(wr + (size_t)o * RED + k0 + kbB);
            bF[ct].u[0] = p[0];
            bF[ct].u[1] = p[2];
        }
        #pragma unroll
        for (int rt = 0; rt < 2; ++rt)
            #pragma unroll
            for (int ct = 0; ct < 2; ++ct)
                acc[rt][ct] = __builtin_amdgcn_wmma_f32_16x16x32_bf16(
                    false, aF[rt].v, false, bF[ct].v,
                    (short)0, acc[rt][ct], false, false);
    }
    __syncthreads();   // everyone done reading As before it becomes Cs

    // ---- spill C tile (+bias) to LDS, per the 16x16 f32 C layout
    #pragma unroll
    for (int rt = 0; rt < 2; ++rt) {
        #pragma unroll
        for (int ct = 0; ct < 2; ++ct) {
            int col = (n0 + ct) * 16 + alane;
            float bv = bias[col];
            #pragma unroll
            for (int r = 0; r < 8; ++r) {
                int row = rt * 16 + ((lane >> 4) << 3) + r;
                Cs[row * H_ + col] = acc[rt][ct][r] + bv;
            }
        }
    }
    __syncthreads();

    // ---- LayerNorm + ReLU + projection; 8 threads per row
    int row = tid >> 3;
    int sub = tid & 7;
    const float* crow = Cs + row * H_;
    float s1 = 0.f, s2 = 0.f;
    for (int j = sub; j < H_; j += 8) { float v = crow[j]; s1 += v; s2 += v * v; }
    for (int off = 4; off; off >>= 1) {
        s1 += __shfl_xor(s1, off, 8);
        s2 += __shfl_xor(s2, off, 8);
    }
    float mu  = s1 * (1.f / H_);
    float var = s2 * (1.f / H_) - mu * mu;
    float rs  = rsqrtf(var + 1e-5f);
    float dot = 0.f;
    for (int j = sub; j < H_; j += 8) {
        float v = (crow[j] - mu) * rs * g[j] + bet[j];
        v = fmaxf(v, 0.f);
        dot += v * lw[j];
    }
    for (int off = 4; off; off >>= 1) dot += __shfl_xor(dot, off, 8);
    if (sub == 0) {
        int t = t0 + row;
        float mv = dot + lb[0];
        m_out[b * T_ + t] = (t < token_nums[b]) ? mv : 0.f;
    }
}

// ---------------------------------------------------------------------------
// Per-batch clipped inclusive scan of durations -> cum[b][t]
__global__ void scan_kernel(const float* __restrict__ ali,
                            const int*   __restrict__ gt,
                            float*       __restrict__ cum)
{
    int b = blockIdx.x, tid = threadIdx.x;
    __shared__ float sd[256];
    float v[4], run = 0.f;
    int base = b * T_ + tid * 4;
    #pragma unroll
    for (int i = 0; i < 4; ++i) { run += ali[base + i]; v[i] = run; }
    float tsum = run;
    sd[tid] = tsum;
    __syncthreads();
    for (int off = 1; off < 256; off <<= 1) {
        float t = 0.f;
        if (tid >= off) t = sd[tid - off];
        __syncthreads();
        sd[tid] += t;
        __syncthreads();
    }
    float excl = sd[tid] - tsum;
    float gmax = (float)gt[b];
    #pragma unroll
    for (int i = 0; i < 4; ++i)
        cum[base + i] = fminf(excl + v[i], gmax);
}

// Per-batch masked exp-sum -> int32
__global__ void expsum_kernel(const float* __restrict__ m,
                              const int*   __restrict__ tn,
                              int*         __restrict__ outi)
{
    int b = blockIdx.x, tid = threadIdx.x;
    __shared__ float sd[256];
    int n = tn[b];
    float s = 0.f;
    for (int t = tid; t < T_; t += 256)
        if (t < n) s += expf(m[b * T_ + t]);
    sd[tid] = s;
    __syncthreads();
    for (int off = 128; off; off >>= 1) {
        if (tid < off) sd[tid] += sd[tid + off];
        __syncthreads();
    }
    if (tid == 0) outi[b] = (int)sd[0];
}

// Duration expansion: per (b, frame) binary-search cum (wave-uniform -> SALU),
// then coalesced 256-float row copy / zero fill.
__global__ void expand_kernel(const float* __restrict__ x,
                              const float* __restrict__ cum,
                              float*       __restrict__ xnew)
{
    int blk = blockIdx.x;
    int b   = blk >> 12;          // TMEL = 4096
    int f   = blk & (TMEL - 1);
    const float* c = cum + b * T_;
    float ff = (float)f;
    int lo = 0, hi = T_;
    while (lo < hi) {             // first idx with cum[idx] > f  == searchsorted right
        int mid = (lo + hi) >> 1;
        if (c[mid] <= ff) lo = mid + 1; else hi = mid;
    }
    int tok = lo;
    float val = 0.f;
    if (tok < T_) val = x[((size_t)b * T_ + tok) * H_ + threadIdx.x];
    xnew[(size_t)blk * H_ + threadIdx.x] = val;
}

// ---------------------------------------------------------------------------
extern "C" void kernel_launch(void* const* d_in, const int* in_sizes, int n_in,
                              void* d_out, int out_size, void* d_ws, size_t ws_size,
                              hipStream_t stream)
{
    const float* x          = (const float*)d_in[0];
    const int*   token_nums = (const int*)  d_in[1];
    const float* ali        = (const float*)d_in[2];
    const int*   gt         = (const int*)  d_in[3];
    // d_in[4..7] = conv1_* / ln1_* : dead in the reference, unused
    const float* conv2_w    = (const float*)d_in[8];
    const float* conv2_b    = (const float*)d_in[9];
    const float* ln2_g      = (const float*)d_in[10];
    const float* ln2_b      = (const float*)d_in[11];
    const float* lin_w      = (const float*)d_in[12];
    const float* lin_b      = (const float*)d_in[13];

    // outputs concatenated: x_new [B,TMEL,H] f32 | m [B,T] f32 | num [B] i32
    float* x_new = (float*)d_out;
    float* m_out = x_new + (size_t)B_ * TMEL * H_;
    int*   n_out = (int*)(m_out + (size_t)B_ * T_);

    // workspace: Wr bf16 (640 KB) | xp bf16 (16.8 MB) | cum f32 (128 KB)
    char* ws = (char*)d_ws;
    unsigned short* wr  = (unsigned short*)ws;
    unsigned short* xpb = (unsigned short*)(ws + (size_t)H_ * RED * 2);
    float*          cum = (float*)(ws + (size_t)H_ * RED * 2 + (size_t)B_ * TP * H_ * 2);

    pack_w<<<RED, 256, 0, stream>>>(conv2_w, wr);
    pack_x<<<(B_ * TP * H_) / 256, 256, 0, stream>>>(x, xpb);
    scan_kernel<<<B_, 256, 0, stream>>>(ali, gt, cum);

    size_t lds_bytes = (size_t)ROWS * RED * 2;   // 80 KB (Cs overlays As)
    conv_ln_kernel<<<B_ * (T_ / ROWS), 256, lds_bytes, stream>>>(
        xpb, wr, conv2_b, ln2_g, ln2_b, lin_w, lin_b, token_nums, m_out);

    expsum_kernel<<<B_, 256, 0, stream>>>(m_out, token_nums, n_out);
    expand_kernel<<<B_ * TMEL, 256, 0, stream>>>(x, cum, x_new);
}
